// Attention_21492016349641
// MI455X (gfx1250) — matrix operand order repaired and checked by reading, not compiled, not run on hardware
//
#include <hip/hip_runtime.h>
#include <math.h>

typedef __attribute__((ext_vector_type(16))) _Float16 v16h;
typedef __attribute__((ext_vector_type(8)))  _Float16 v8h;
typedef __attribute__((ext_vector_type(8)))  float    v8f;

#define BATCH 64
#define SEQ   2048
#define ENC   512
#define ATT   512
#define STILE 128          // seq rows staged per block
#define LDPAD 520          // halves per LDS row (512 + 8 pad, keeps 16B alignment)

// Branchless tanh: tanh(x) = 1 - 2/(exp(2x)+1).
// exp(2x)->inf => rcp->0 => +1 ; exp(2x)->0 => -1. No EXEC divergence.
__device__ __forceinline__ float fast_tanhf(float x) {
    float e2 = __expf(2.0f * x);                  // v_mul + v_exp_f32
    float r  = __builtin_amdgcn_rcpf(e2 + 1.0f);  // v_rcp_f32
    return fmaf(-2.0f, r, 1.0f);
}

// ---------------------------------------------------------------------------
// Kernel 1: attn_1[b,a] = sum_d h[b,d] * W_dec[a,d]    (64 x 512, tiny)
// ---------------------------------------------------------------------------
__global__ __launch_bounds__(256)
void attn1_kernel(const float* __restrict__ h,
                  const float* __restrict__ Wdec,
                  float* __restrict__ attn1) {
    __shared__ float hrow[ENC];
    const int b = blockIdx.x;
    const int t = threadIdx.x;
    for (int i = t; i < ENC; i += 256) hrow[i] = h[(size_t)b * ENC + i];
    __syncthreads();
    for (int a = t; a < ATT; a += 256) {
        const float* w = Wdec + (size_t)a * ENC;
        float s = 0.f;
        #pragma unroll 8
        for (int d = 0; d < ENC; ++d) s = fmaf(hrow[d], w[d], s);
        attn1[(size_t)b * ATT + a] = s;
    }
}

// ---------------------------------------------------------------------------
// Kernel 2: convert W_enc (a,e) f32 -> f16 (L2-resident B operand, 512 KB)
// ---------------------------------------------------------------------------
__global__ __launch_bounds__(256)
void cvt16_kernel(const float* __restrict__ src, _Float16* __restrict__ dst, int n) {
    int i = (blockIdx.x * 256 + threadIdx.x) * 4;
    if (i < n) {
        float4 v = *(const float4*)(src + i);
        dst[i + 0] = (_Float16)v.x;
        dst[i + 1] = (_Float16)v.y;
        dst[i + 2] = (_Float16)v.z;
        dst[i + 3] = (_Float16)v.w;
    }
}

// ---------------------------------------------------------------------------
// Kernel 3: fused scores[b,s] = sum_a tanh(attn1[b,a] + (V·W_encT)[b,s,a]) * w[a]
// Block = (b, s-tile of 128). 8 waves, each owns 16 seq rows.
// GEMM via v_wmma_f32_16x16x32_f16; two N-tiles in flight (two independent
// accumulator chains, A fragment reused 2x).
// ---------------------------------------------------------------------------
__global__ __launch_bounds__(256)
void scores_kernel(const float* __restrict__ V,
                   const _Float16* __restrict__ Wenc16,
                   const float* __restrict__ attn1,
                   const float* __restrict__ wfull,
                   float* __restrict__ scores) {
    __shared__ _Float16 tileV[STILE * LDPAD];   // 133 KB (WGP has 320 KB)
    __shared__ float a1[ATT];
    __shared__ float wf[ATT];

    const int b  = blockIdx.x;
    const int s0 = blockIdx.y * STILE;
    const int t  = threadIdx.x;

    // ---- stage V[b, s0:s0+128, :] -> f16 LDS (uniform trip count: no EXEC holes)
    const float* Vb = V + ((size_t)b * SEQ + s0) * ENC;
    for (int i = t; i < STILE * ENC / 4; i += 256) {
        int row = i >> 7;              // 128 float4 per row
        int c4  = i & 127;
        float4 v = *(const float4*)(Vb + (size_t)row * ENC + c4 * 4);
        _Float16* d = &tileV[row * LDPAD + c4 * 4];
        d[0] = (_Float16)v.x; d[1] = (_Float16)v.y;
        d[2] = (_Float16)v.z; d[3] = (_Float16)v.w;
    }
    for (int i = t; i < ATT; i += 256) {
        a1[i] = attn1[(size_t)b * ATT + i];
        wf[i] = wfull[i];
    }
    __syncthreads();

    const int wid  = t >> 5;
    const int lane = t & 31;
    const int half = lane >> 4;        // 0: lanes 0-15, 1: lanes 16-31
    const int ln   = lane & 15;
    const int rowA = wid * 16 + ln;    // A-fragment row this lane supplies

    float sc[8];
    #pragma unroll
    for (int i = 0; i < 8; ++i) sc[i] = 0.f;

    const _Float16* aRow = &tileV[rowA * LDPAD + half * 16];

    for (int nt = 0; nt < ATT / 16; nt += 2) {       // 16 passes, 2 N-tiles each
        v8f acc0 = {};
        v8f acc1 = {};
        const _Float16* b0p = Wenc16 + (size_t)(nt * 16 + ln) * ENC + half * 16;
        const _Float16* b1p = b0p + (size_t)16 * ENC;
        #pragma unroll 4
        for (int kt = 0; kt < ENC / 32; ++kt) {      // 16 K-steps of 32
            const int kb = kt * 32;
            // A 16x32 f16: lane-half h holds K = kb+16h+{0..15} (matches B)
            v8h alo = *(const v8h*)(aRow + kb);        // K = kb+16h .. +7
            v8h ahi = *(const v8h*)(aRow + kb + 8);    // K = kb+16h+8 .. +15
            v16h afr = __builtin_shufflevector(alo, ahi,
                        0,1,2,3,4,5,6,7,8,9,10,11,12,13,14,15);
            // B 32x16 f16: lane column N=ln, K = kb + 16h + 0..15 (32B aligned)
            v16h bf0 = *(const v16h*)(b0p + kb);
            v16h bf1 = *(const v16h*)(b1p + kb);
            acc0 = __builtin_amdgcn_wmma_f32_16x16x32_f16(
                       false, afr, false, bf0, (short)0, acc0, false, false);
            acc1 = __builtin_amdgcn_wmma_f32_16x16x32_f16(
                       false, afr, false, bf1, (short)0, acc1, false, false);
        }
        // C layout: VGPR i -> (M = i + 8*half, N = ln); apply tanh * w, sum over a
        const int a0 = nt * 16 + ln;
        const int a1i = a0 + 16;
        const float add0 = a1[a0],  w0 = wf[a0];
        const float add1 = a1[a1i], w1 = wf[a1i];
        #pragma unroll
        for (int i = 0; i < 8; ++i) {
            sc[i] = fmaf(w0, fast_tanhf(acc0[i] + add0), sc[i]);
            sc[i] = fmaf(w1, fast_tanhf(acc1[i] + add1), sc[i]);
        }
    }

    // reduce the 16 lanes of each half (the N dimension of the C tile)
    #pragma unroll
    for (int i = 0; i < 8; ++i) {
        float s = sc[i];
        s += __shfl_xor(s, 1, 32);
        s += __shfl_xor(s, 2, 32);
        s += __shfl_xor(s, 4, 32);
        s += __shfl_xor(s, 8, 32);
        sc[i] = s;
    }
    if (ln == 0) {                     // lane 0 -> rows 0..7, lane 16 -> rows 8..15
        #pragma unroll
        for (int i = 0; i < 8; ++i) {
            int m = i + half * 8;
            scores[(size_t)b * SEQ + s0 + wid * 16 + m] = sc[i];
        }
    }
}

// ---------------------------------------------------------------------------
// Kernel 4: softmax over seq + context[b,e] = sum_s alpha[b,s] * V[b,s,e]
// One block per batch row; coalesced float2 per thread over e.
// ---------------------------------------------------------------------------
__global__ __launch_bounds__(256)
void softmax_ctx_kernel(const float* __restrict__ V,
                        const float* __restrict__ scores,
                        float* __restrict__ out) {
    __shared__ float sAlpha[SEQ];      // 8 KB
    __shared__ float red[8];
    const int b    = blockIdx.x;
    const int t    = threadIdx.x;
    const int lane = t & 31;
    const int wid  = t >> 5;

    const float* sb = scores + (size_t)b * SEQ;
    float loc[8];
    float mx = -INFINITY;
    #pragma unroll
    for (int k = 0; k < 8; ++k) {
        loc[k] = sb[t + k * 256];
        mx = fmaxf(mx, loc[k]);
    }
    for (int m = 16; m >= 1; m >>= 1) mx = fmaxf(mx, __shfl_xor(mx, m, 32));
    if (lane == 0) red[wid] = mx;
    __syncthreads();
    mx = red[0];
    #pragma unroll
    for (int i = 1; i < 8; ++i) mx = fmaxf(mx, red[i]);

    float sum = 0.f;
    #pragma unroll
    for (int k = 0; k < 8; ++k) {
        float e = expf(loc[k] - mx);
        sAlpha[t + k * 256] = e;
        sum += e;
    }
    for (int m = 16; m >= 1; m >>= 1) sum += __shfl_xor(sum, m, 32);
    __syncthreads();                   // all max-reads of red[] done
    if (lane == 0) red[wid] = sum;
    __syncthreads();
    sum = 0.f;
    #pragma unroll
    for (int i = 0; i < 8; ++i) sum += red[i];
    const float inv = 1.0f / sum;

    const float* Vb = V + (size_t)b * SEQ * ENC;
    const int e0 = t * 2;
    float c0 = 0.f, c1 = 0.f;
    #pragma unroll 4
    for (int s = 0; s < SEQ; ++s) {
        float a = sAlpha[s] * inv;     // LDS broadcast
        float2 v = *(const float2*)(Vb + (size_t)s * ENC + e0);
        c0 = fmaf(a, v.x, c0);
        c1 = fmaf(a, v.y, c1);
    }
    out[(size_t)b * ENC + e0]     = c0;
    out[(size_t)b * ENC + e0 + 1] = c1;
}

// ---------------------------------------------------------------------------
extern "C" void kernel_launch(void* const* d_in, const int* in_sizes, int n_in,
                              void* d_out, int out_size, void* d_ws, size_t ws_size,
                              hipStream_t stream) {
    (void)in_sizes; (void)n_in; (void)out_size; (void)ws_size;
    const float* h     = (const float*)d_in[0];   // (64, 512)
    const float* V     = (const float*)d_in[1];   // (64, 2048, 512)
    const float* Wdec  = (const float*)d_in[2];   // (512, 512)
    const float* Wenc  = (const float*)d_in[3];   // (512, 512)
    const float* wfull = (const float*)d_in[4];   // (512,)
    float* out = (float*)d_out;                   // (64, 512)

    // workspace layout (~1.2 MB total)
    char* ws = (char*)d_ws;
    float*    scores = (float*)ws;                              // 64*2048*4 = 524288 B
    float*    attn1  = (float*)(ws + 524288);                   // 64*512*4  = 131072 B
    _Float16* Wenc16 = (_Float16*)(ws + 524288 + 131072);       // 512*512*2 = 524288 B

    attn1_kernel<<<BATCH, 256, 0, stream>>>(h, Wdec, attn1);
    cvt16_kernel<<<(ATT * ENC) / (256 * 4), 256, 0, stream>>>(Wenc, Wenc16, ATT * ENC);

    dim3 g(BATCH, SEQ / STILE);                  // 64 x 16 = 1024 blocks
    scores_kernel<<<g, 256, 0, stream>>>(V, Wenc16, attn1, wfull, scores);

    softmax_ctx_kernel<<<BATCH, 256, 0, stream>>>(V, scores, out);
}
